// LSTM2_65901978190233
// MI455X (gfx1250) — compile-verified, matrix operand order checked by reading, not run on hardware
//
#include <hip/hip_runtime.h>
#include <hip/hip_bf16.h>
#include <cstdint>
#include <cstddef>

// ---------------------------------------------------------------------------
// Two independent LSTMs w/ projection, persistent-kernel, gate-split design.
//   lstm0: inp -> out2   (second tuple element of d_out)
//   lstm1: out -> inp2   (first tuple element of d_out)
// Grid = 16 WGs (2 LSTMs x 8 gate-slices), 512 threads (16 wave32) each.
// Weights resident in 313KB LDS per WGP; cell state + projection weights +
// bias in VGPRs; per-step cross-WG reduction via L2 + release/acquire ctrs.
// Latency hiding relies on 4 waves/SIMD coresidency -> keep VGPRs moderate.
// ---------------------------------------------------------------------------

typedef __attribute__((ext_vector_type(16))) _Float16 v16h;
typedef __attribute__((ext_vector_type(8)))  _Float16 v8h;
typedef __attribute__((ext_vector_type(8)))  float    v8f;
typedef __attribute__((ext_vector_type(4)))  float    v4f;

#define B_      64
#define T_      512
#define DIN     256
#define HID     512
#define PRJ     256
#define KTOT    512            // DIN + PRJ
#define NWG     8              // workgroups per LSTM (gate split)
#define HU      64             // hidden units per WG
#define THREADS 512
#define NBLOCKS 16

// ---- LDS layout (units: halves unless noted) ----
#define WROW 520               // 512 + 8 pad  -> conflict-free B-operand reads
#define RROW 72                // 64 + 8 pad
#define SROW 72
#define LDSW_OFF 0
#define LDSR_OFF (256 * WROW)
#define SBUF_OFF (LDSR_OFF + 256 * RROW)
#define BIAS_OFF_BYTES ((SBUF_OFF + 64 * SROW) * 2)
#define SMEM_BYTES (BIAS_OFF_BYTES + 256 * 4)   // = 313,344 <= 320KB

// ---- workspace layout (bytes) ----
#define WS_SYNC_OFF   0         // 16KB of u32 counters (memset to 0 each launch)
#define WS_HBUF_OFF   16384     // 2 * 64 * 256 f16 = 64KB (memset to 0)
#define WS_PART_OFF   81920     // 2 * 8 * 64 * 256 f32 = 1MB
#define WS_XF_OFF     1130496   // 2 * T*B*DIN f16 = 32MB
#define WS_ZERO_BYTES 81920

__device__ __forceinline__ float sigf(float x) {
  return __builtin_amdgcn_rcpf(1.0f + __expf(-x));
}
__device__ __forceinline__ float tanhfast(float x) {
  return 2.0f * __builtin_amdgcn_rcpf(1.0f + __expf(-2.0f * x)) - 1.0f;
}

union AV { v16h v; v8h h[2]; };

__global__ __launch_bounds__(THREADS, 1)
void lstm2_persistent_kernel(
    const float* __restrict__ inp,  const float* __restrict__ outp,
    const float* __restrict__ wih1, const float* __restrict__ whh1,
    const float* __restrict__ b1,   const float* __restrict__ whr1,
    const float* __restrict__ wih2, const float* __restrict__ whh2,
    const float* __restrict__ b2,   const float* __restrict__ whr2,
    float* __restrict__ out, unsigned char* __restrict__ ws)
{
  extern __shared__ char smem_raw[];
  _Float16* lds   = reinterpret_cast<_Float16*>(smem_raw);
  float*    biasL = reinterpret_cast<float*>(smem_raw + BIAS_OFF_BYTES);

  const int tid  = threadIdx.x;
  const int wg   = blockIdx.x;
  const int lstm = wg >> 3;
  const int j    = wg & 7;          // gate-slice id within this LSTM
  const int hu0  = j * HU;
  const int wave = tid >> 5;
  const int lane = tid & 31;
  const int lm   = lane & 15;       // row/col within tile
  const int lh   = lane >> 4;       // K-half select per WMMA operand layout

  unsigned int* sync = reinterpret_cast<unsigned int*>(ws + WS_SYNC_OFF);
  _Float16* hb_all   = reinterpret_cast<_Float16*>(ws + WS_HBUF_OFF);
  float*    part_all = reinterpret_cast<float*>(ws + WS_PART_OFF);
  _Float16* xf_all   = reinterpret_cast<_Float16*>(ws + WS_XF_OFF);

  const float* wih = lstm ? wih2 : wih1;
  const float* whh = lstm ? whh2 : whh1;
  const float* bia = lstm ? b2   : b1;
  const float* whr = lstm ? whr2 : whr1;
  _Float16* xf   = xf_all + (size_t)lstm * T_ * B_ * DIN;  // [T][B][DIN] f16
  _Float16* hb   = hb_all + lstm * B_ * PRJ;               // [B][PRJ] f16
  float*    part = part_all + (lstm * NWG + j) * B_ * PRJ; // this WG's partial
  float*    outbase = out + (lstm == 0 ? (size_t)B_ * T_ * PRJ : (size_t)0);

  // ---- Phase 0a: convert + transpose inputs -> f16 [T][B][DIN] (all WGs) ----
  {
    const int total4 = 2 * T_ * B_ * (DIN / 4);
    const int stride = gridDim.x * THREADS;
    for (int i = blockIdx.x * THREADS + tid; i < total4; i += stride) {
      int c4 = i & 63;  int r = i >> 6;
      int b  = r & 63;  r >>= 6;
      int t  = r & 511; int l = r >> 9;
      const float* s = (l ? outp : inp) + ((size_t)b * T_ + t) * DIN + c4 * 4;
      v4f v = *reinterpret_cast<const v4f*>(s);
      _Float16* d = xf_all + (size_t)l * T_ * B_ * DIN
                  + ((size_t)t * B_ + b) * DIN + c4 * 4;
      d[0] = (_Float16)v.x; d[1] = (_Float16)v.y;
      d[2] = (_Float16)v.z; d[3] = (_Float16)v.w;
    }
  }

  // ---- Phase 0b: repack this WG's weight slice into LDS (f32 -> f16) ----
  // gate rows r=0..255 map to grow = gate_section*HID + hu0 + (r&63)
  for (int e = tid; e < 256 * KTOT; e += THREADS) {
    int r = e >> 9, c = e & 511;
    int grow = (r >> 6) * HID + hu0 + (r & 63);
    float w = (c < DIN) ? wih[(size_t)grow * DIN + c]
                        : whh[(size_t)grow * PRJ + (c - DIN)];
    lds[LDSW_OFF + r * WROW + c] = (_Float16)w;
  }
  for (int e = tid; e < PRJ * HU; e += THREADS) {
    int p = e >> 6, hh = e & 63;
    lds[LDSR_OFF + p * RROW + hh] = (_Float16)whr[(size_t)p * HID + hu0 + hh];
  }
  for (int r = tid; r < 256; r += THREADS)
    biasL[r] = bia[(r >> 6) * HID + hu0 + (r & 63)];
  __syncthreads();

  // ---- One-time grid barrier: x-conversion visible everywhere ----
  if (tid == 0) {
    __hip_atomic_fetch_add(&sync[0], 1u, __ATOMIC_RELEASE, __HIP_MEMORY_SCOPE_AGENT);
    while (__hip_atomic_load(&sync[0], __ATOMIC_ACQUIRE, __HIP_MEMORY_SCOPE_AGENT)
           < (unsigned)NBLOCKS)
      __builtin_amdgcn_s_sleep(1);
  }
  __syncthreads();

  // step-indexed counters: A = partials ready, B = h(t) ready
  #define CTRA(t) sync[8 + ((lstm) * T_ + (t)) * 2]
  #define CTRB(t) sync[9 + ((lstm) * T_ + (t)) * 2]

  const int m = wave >> 2;     // batch tile  (0..3)  : rows m*16..m*16+15
  const int n = wave & 3;      // hidden tile (0..3)  : 16 of this WG's 64 HUs
  v8f cst = {};                // persistent cell-state tile [16b x 16h]

  // ---- t-invariant values hoisted into registers for the whole scan ----
  const float bv0 = biasL[0 * 64 + n * 16 + lm];
  const float bv1 = biasL[1 * 64 + n * 16 + lm];
  const float bv2 = biasL[2 * 64 + n * 16 + lm];
  const float bv3 = biasL[3 * 64 + n * 16 + lm];
  // projection weight tiles (w_hr slice): 4 q-tiles x 2 K-tiles, 64 VGPRs
  const int pg = wave & 3;
  AV rw[4][2];
  #pragma unroll
  for (int q = 0; q < 4; ++q) {
    const _Float16* bp = lds + LDSR_OFF + ((pg * 4 + q) * 16 + lm) * RROW + lh * 8;
    #pragma unroll
    for (int k2 = 0; k2 < 2; ++k2) {
      rw[q][k2].h[0] = *reinterpret_cast<const v8h*>(bp + k2 * 32);
      rw[q][k2].h[1] = *reinterpret_cast<const v8h*>(bp + k2 * 32 + 16);
    }
  }
  // per-wave invariant LDS base for gate-weight B-tiles
  const _Float16* wbase = lds + LDSW_OFF + (n * 16 + lm) * WROW + lh * 8;

  for (int t = 0; t < T_; ++t) {
    // ---- wait until h(t-1) is published by all 8 reducers ----
    if (t > 0) {
      if (tid == 0) {
        while (__hip_atomic_load(&CTRB(t - 1), __ATOMIC_ACQUIRE,
                                 __HIP_MEMORY_SCOPE_AGENT) < (unsigned)NWG)
          __builtin_amdgcn_s_sleep(1);
      }
      __syncthreads();   // also guards LDS s-buffer reuse
    }

    // ---- gates = [x_t | h] @ Wslice^T  (f16 WMMA, f32 accum, inline-0 C) ----
    const _Float16* xrow = xf + ((size_t)t * B_ + m * 16 + lm) * DIN;
    const _Float16* hrow = hb + (m * 16 + lm) * PRJ;
    if (t + 1 < T_) __builtin_prefetch(xrow + B_ * DIN, 0, 1);  // next-step x

    v8f acc[4];
    #pragma unroll
    for (int kt = 0; kt < 16; ++kt) {
      // A tile: batch rows x K-slice (x_t for kt<8, h(t-1) for kt>=8)
      AV a;
      const _Float16* ap = (kt < 8) ? (xrow + kt * 32 + lh * 8)
                                    : (hrow + (kt - 8) * 32 + lh * 8);
      a.h[0] = *reinterpret_cast<const v8h*>(ap);
      a.h[1] = *reinterpret_cast<const v8h*>(ap + 16);

      // all four gate-section B-tiles, then the four WMMAs
      AV bm[4];
      #pragma unroll
      for (int s = 0; s < 4; ++s) {
        const _Float16* bp = wbase + (s * 64) * WROW + kt * 32;
        bm[s].h[0] = *reinterpret_cast<const v8h*>(bp);
        bm[s].h[1] = *reinterpret_cast<const v8h*>(bp + 16);
      }
      if (kt == 0) {
        v8f z = {};
        #pragma unroll
        for (int s = 0; s < 4; ++s)
          acc[s] = __builtin_amdgcn_wmma_f32_16x16x32_f16(
              false, a.v, false, bm[s].v, (short)0, z, false, false);
      } else {
        #pragma unroll
        for (int s = 0; s < 4; ++s)
          acc[s] = __builtin_amdgcn_wmma_f32_16x16x32_f16(
              false, a.v, false, bm[s].v, (short)0, acc[s], false, false);
      }
    }

    // ---- elementwise cell update (bias folded here; c stays in VGPRs) ----
    #pragma unroll
    for (int r = 0; r < 8; ++r) {
      float iv = sigf(acc[0][r] + bv0);
      float fv = sigf(acc[1][r] + bv1);
      float gv = tanhfast(acc[2][r] + bv2);
      float ov = sigf(acc[3][r] + bv3);
      float cn = fv * cst[r] + iv * gv;
      cst[r] = cn;
      float sv = ov * tanhfast(cn);
      int brow = m * 16 + r + lh * 8;                // C/D layout: M = r (+8)
      lds[SBUF_OFF + brow * SROW + n * 16 + lm] = (_Float16)sv;
    }
    __syncthreads();

    // ---- partial projection: part[64,256] = s[64,64slice] @ whr_slice^T ----
    {
      const int m2 = wave >> 2;
      AV a0, a1;
      const _Float16* ap0 = lds + SBUF_OFF + (m2 * 16 + lm) * SROW + lh * 8;
      a0.h[0] = *reinterpret_cast<const v8h*>(ap0);
      a0.h[1] = *reinterpret_cast<const v8h*>(ap0 + 16);
      a1.h[0] = *reinterpret_cast<const v8h*>(ap0 + 32);
      a1.h[1] = *reinterpret_cast<const v8h*>(ap0 + 48);
      #pragma unroll
      for (int q = 0; q < 4; ++q) {
        v8f z = {};
        v8f pa = __builtin_amdgcn_wmma_f32_16x16x32_f16(
            false, a0.v, false, rw[q][0].v, (short)0, z, false, false);
        pa = __builtin_amdgcn_wmma_f32_16x16x32_f16(
            false, a1.v, false, rw[q][1].v, (short)0, pa, false, false);
        #pragma unroll
        for (int r = 0; r < 8; ++r)
          part[(m2 * 16 + r + lh * 8) * PRJ + (pg * 4 + q) * 16 + lm] = pa[r];
      }
    }
    __threadfence();        // each thread drains its own stores to L2
    __syncthreads();
    if (tid == 0)
      __hip_atomic_fetch_add(&CTRA(t), 1u, __ATOMIC_RELEASE, __HIP_MEMORY_SCOPE_AGENT);

    // ---- wait for all 8 partials, then reduce our 8 batch rows ----
    if (tid == 0) {
      while (__hip_atomic_load(&CTRA(t), __ATOMIC_ACQUIRE,
                               __HIP_MEMORY_SCOPE_AGENT) < (unsigned)NWG)
        __builtin_amdgcn_s_sleep(1);
    }
    __syncthreads();
    {
      int b  = j * 8 + (tid >> 6);
      int p4 = (tid & 63) * 4;
      const float* pb = part_all + (size_t)lstm * NWG * B_ * PRJ + b * PRJ + p4;
      v4f s = {};
      #pragma unroll
      for (int w = 0; w < NWG; ++w)
        s += *reinterpret_cast<const v4f*>(pb + (size_t)w * B_ * PRJ);
      *reinterpret_cast<v4f*>(outbase + ((size_t)b * T_ + t) * PRJ + p4) = s;
      _Float16* hd = hb + b * PRJ + p4;
      hd[0] = (_Float16)s.x; hd[1] = (_Float16)s.y;
      hd[2] = (_Float16)s.z; hd[3] = (_Float16)s.w;
    }
    __threadfence();
    __syncthreads();
    if (tid == 0)
      __hip_atomic_fetch_add(&CTRB(t), 1u, __ATOMIC_RELEASE, __HIP_MEMORY_SCOPE_AGENT);
  }
  #undef CTRA
  #undef CTRB
}

extern "C" void kernel_launch(void* const* d_in, const int* in_sizes, int n_in,
                              void* d_out, int out_size, void* d_ws, size_t ws_size,
                              hipStream_t stream) {
  (void)in_sizes; (void)n_in; (void)out_size; (void)ws_size;
  const float* inp  = (const float*)d_in[0];
  const float* outp = (const float*)d_in[1];
  const float* wih1 = (const float*)d_in[2];
  const float* whh1 = (const float*)d_in[3];
  const float* b1   = (const float*)d_in[4];
  const float* whr1 = (const float*)d_in[5];
  const float* wih2 = (const float*)d_in[6];
  const float* whh2 = (const float*)d_in[7];
  const float* b2   = (const float*)d_in[8];
  const float* whr2 = (const float*)d_in[9];

  // allow 313KB dynamic LDS (one WG per WGP); harmless if already set
  hipFuncSetAttribute((const void*)lstm2_persistent_kernel,
                      hipFuncAttributeMaxDynamicSharedMemorySize, SMEM_BYTES);

  // zero sync counters + initial h state (graph-capturable, replayed each run)
  hipMemsetAsync(d_ws, 0, WS_ZERO_BYTES, stream);

  lstm2_persistent_kernel<<<NBLOCKS, THREADS, SMEM_BYTES, stream>>>(
      inp, outp, wih1, whh1, b1, whr1, wih2, whh2, b2, whr2,
      (float*)d_out, (unsigned char*)d_ws);
}
